// RetinaNetHeads_56212531970623
// MI455X (gfx1250) — matrix operand order repaired and checked by reading, not compiled, not run on hardware
//
#include <hip/hip_runtime.h>
#include <hip/hip_bf16.h>

typedef __attribute__((ext_vector_type(16))) _Float16 v16h;
typedef __attribute__((ext_vector_type(8)))  float    v8f;
typedef __attribute__((ext_vector_type(8)))  unsigned v8u;
typedef __attribute__((ext_vector_type(4)))  unsigned v4u;

#define CIN 256
#define CAP 16384
#define NPOOL 5000
#define BBOX_CLIP 4.135166556742356f

// ---------------------------------------------------------------------------
// f32 -> f16 convert (features)
__global__ void k_cvt_f16(const float* __restrict__ x, _Float16* __restrict__ y, int n) {
  for (int i = blockIdx.x * blockDim.x + threadIdx.x; i < n; i += gridDim.x * blockDim.x)
    y[i] = (_Float16)x[i];
}

// pack weights: in [rows][256][3][3] f32 -> out [rows][9][256] f16
__global__ void k_pack_w(const float* __restrict__ w, _Float16* __restrict__ out, int rows) {
  int total = rows * 2304;
  for (int i = blockIdx.x * blockDim.x + threadIdx.x; i < total; i += gridDim.x * blockDim.x) {
    int co = i / 2304, r = i - co * 2304;
    int tap = r >> 8, ci = r & 255;
    out[i] = (_Float16)w[(co * 256 + ci) * 9 + tap];
  }
}

__global__ void k_zero_counters(int* c) { if (threadIdx.x < 8) c[threadIdx.x] = 0; }

// ---------------------------------------------------------------------------
// 3x3 same-pad conv as implicit GEMM with WMMA f16->f32.
// X: [256][HW] f16, Wp: [Cout][9][256] f16, bias[Cout] f32.
// Output: relu? f16 Yh[co][HW] : f32 Yf[co][HW].
// Block: 256 thr = 8 waves -> 32(co) x 64(px) tile; wave = 16x16 sub-tile.
// A tile is DMA'd into LDS by the Tensor Data Mover (tensor_load_to_lds);
// B tile (im2col with zero-pad borders) is gathered by the vector pipe.
__global__ __launch_bounds__(256)
void k_conv3x3_wmma(const _Float16* __restrict__ X,
                    const _Float16* __restrict__ Wp,
                    const float* __restrict__ bias,
                    _Float16* __restrict__ Yh,
                    float* __restrict__ Yf,
                    int H, int wsh, int Cout, int relu) {
  __shared__ _Float16 As[32][264];   // co x ci (per tap), 8-half row pad
  __shared__ _Float16 Bs[64][264];   // px x ci (per tap)
  const int W = 1 << wsh;
  const int HW = H * W;
  const int tid  = threadIdx.x;
  const int lane = tid & 31;
  const int wv   = tid >> 5;
  const int mt   = wv >> 2;          // 0..1  (co 16-tile)
  const int nt   = wv & 3;           // 0..3  (px 16-tile)
  const int p_base  = blockIdx.x * 64;
  const int co_base = blockIdx.y * 32;

  v8f acc = {};

  const int arow = mt * 16 + (lane & 15);
  const int prow = nt * 16 + (lane & 15);
  const int hsel = lane >> 4;

  // Invariant pieces of the tensor-DMA descriptor (D#).
  const unsigned lds_a = (unsigned)(uintptr_t)&As[0][0];
  const int rows_rem = Cout - co_base;           // OOB rows auto-zero via tensor_dim1

  for (int tap = 0; tap < 9; ++tap) {
    const int ky = tap / 3 - 1, kx = tap % 3 - 1;

    // ---- A tile via TDM: 32 co-rows x 256 ci f16, row stride 2304 elements,
    //      LDS pad 4 DWORDs every 128 DWORDs (-> As row stride 264 halves).
    if (wv == 0) {
      unsigned long long ga =
          (unsigned long long)(uintptr_t)(Wp + ((size_t)(co_base * 9 + tap) << 8));
      v4u g0;
      g0[0] = 1u;                                             // count=1, user mode
      g0[1] = lds_a;                                          // lds_addr
      g0[2] = (unsigned)ga;                                   // global_addr[31:0]
      g0[3] = (unsigned)((ga >> 32) & 0x01FFFFFFu) | (2u << 30); // [56:32] | type=2
      v8u g1;
      g1[0] = (1u << 16)        // data_size = 2 bytes
            | (1u << 20)        // pad_enable
            | (6u << 22)        // pad_interval: 128 DWORDs
            | (3u << 25);       // pad_amount: 4 DWORDs
      g1[1] = (256u & 0xFFFFu) << 16;                         // tensor_dim0 lo16
      g1[2] = ((unsigned)rows_rem & 0xFFFFu) << 16;           // td0 hi=0 | tensor_dim1 lo16
      g1[3] = ((unsigned)rows_rem >> 16) | (256u << 16);      // td1 hi | tile_dim0=256
      g1[4] = 32u;                                            // tile_dim1=32, tile_dim2=0
      g1[5] = 2304u;                                          // tensor_dim0_stride lo32
      g1[6] = 0u;                                             // stride hi | dim1_stride lo
      g1[7] = 0u;
      asm volatile("tensor_load_to_lds %0, %1" :: "s"(g0), "s"(g1) : "memory");
    }

    // ---- B tile: im2col column block, Bs[p][ci] (zero-padded borders)
    for (int idx = tid; idx < 256 * 64; idx += 256) {
      int ci = idx >> 6, p = idx & 63;
      int P = p_base + p;
      _Float16 v = (_Float16)0.0f;
      if (P < HW) {
        int y = P >> wsh, x = P & (W - 1);
        int qy = y + ky, qx = x + kx;
        if ((unsigned)qy < (unsigned)H && (unsigned)qx < (unsigned)W)
          v = X[ci * HW + (qy << wsh) + qx];
      }
      Bs[p][ci] = v;
    }

    if (wv == 0) __builtin_amdgcn_s_wait_tensorcnt(0);
    __syncthreads();

    // 8 WMMA K-steps (K=32 each) covering ci = 0..255 for this tap
    for (int kc = 0; kc < 8; ++kc) {
      const int kb = kc << 5;
      v8u au, bu;
#pragma unroll
      for (int j = 0; j < 8; ++j) {
        // A 16x32 f16 layout: K = (j<4?0:16) + 8*half + 2*(j&3)
        int ka = kb + ((j & 4) << 2) + (hsel << 3) + ((j & 3) << 1);
        au[j] = *(const unsigned*)&As[arow][ka];
        // B 32x16 f16 layout: K = (j<4?0:16) + 8*half + 2*(j&3)
        int kbi = kb + ((j & 4) << 2) + (hsel << 3) + ((j & 3) << 1);
        bu[j] = *(const unsigned*)&Bs[prow][kbi];
      }
      v16h a = __builtin_bit_cast(v16h, au);
      v16h b = __builtin_bit_cast(v16h, bu);
      acc = __builtin_amdgcn_wmma_f32_16x16x32_f16(false, a, false, b,
                                                   (short)0, acc, false, false);
    }
    __syncthreads();
  }

  // epilogue: C/D layout -> VGPR r: M = r + 8*(lane>=16), N = lane%16
  const int P = p_base + nt * 16 + (lane & 15);
#pragma unroll
  for (int r = 0; r < 8; ++r) {
    int co = co_base + mt * 16 + (hsel << 3) + r;
    if (co < Cout && P < HW) {
      float v = acc[r] + bias[co];
      if (relu) v = fmaxf(v, 0.0f);
      if (relu) Yh[co * HW + P] = (_Float16)v;
      else      Yf[co * HW + P] = v;
    }
  }
}

// ---------------------------------------------------------------------------
// sigmoid + threshold + decode -> per-level candidate compaction
__global__ void k_candidates(const float* __restrict__ logits,  // [819][HW]
                             const float* __restrict__ reg,     // [36][HW]
                             const float* __restrict__ anchors, // [Atot][4]
                             int HW, int a_off, int lvl,
                             int* __restrict__ counters,
                             float* __restrict__ cscore,
                             float* __restrict__ cbox,
                             int* __restrict__ clabel) {
  const int total = HW * 819;
  for (int i = blockIdx.x * blockDim.x + threadIdx.x; i < total; i += gridDim.x * blockDim.x) {
    int p = i / 819, rem = i - p * 819;
    int a = rem / 91, c = rem - a * 91;
    float lg = logits[(a * 91 + c) * HW + p];
    float s = 1.0f / (1.0f + expf(-lg));
    if (s > 0.05f) {
      int slot = atomicAdd(&counters[lvl], 1);
      if (slot < CAP) {
        int g = a_off + p * 9 + a;
        float ax1 = anchors[g * 4 + 0], ay1 = anchors[g * 4 + 1];
        float ax2 = anchors[g * 4 + 2], ay2 = anchors[g * 4 + 3];
        float aw = ax2 - ax1, ah = ay2 - ay1;
        float acx = ax1 + 0.5f * aw, acy = ay1 + 0.5f * ah;
        float dx = reg[(a * 4 + 0) * HW + p];
        float dy = reg[(a * 4 + 1) * HW + p];
        float dw = fminf(reg[(a * 4 + 2) * HW + p], BBOX_CLIP);
        float dh = fminf(reg[(a * 4 + 3) * HW + p], BBOX_CLIP);
        float pcx = dx * aw + acx, pcy = dy * ah + acy;
        float pw = expf(dw) * aw, ph = expf(dh) * ah;
        float x1 = fminf(fmaxf(pcx - 0.5f * pw, 0.0f), 512.0f);
        float y1 = fminf(fmaxf(pcy - 0.5f * ph, 0.0f), 512.0f);
        float x2 = fminf(fmaxf(pcx + 0.5f * pw, 0.0f), 512.0f);
        float y2 = fminf(fmaxf(pcy + 0.5f * ph, 0.0f), 512.0f);
        int base = lvl * CAP + slot;
        cscore[base] = s;
        cbox[base * 4 + 0] = x1; cbox[base * 4 + 1] = y1;
        cbox[base * 4 + 2] = x2; cbox[base * 4 + 3] = y2;
        clabel[base] = c;
      }
    }
  }
}

// per-level top-1000 selection into fixed pool (1000 slots/level, -1 fill)
__global__ __launch_bounds__(256)
void k_select(const int* __restrict__ counters,
              float* __restrict__ cscore, const float* __restrict__ cbox,
              const int* __restrict__ clabel,
              float* __restrict__ pscore, float* __restrict__ pbox,
              int* __restrict__ plabel) {
  __shared__ float sv[256];
  __shared__ int   si[256];
  const int l = blockIdx.x;
  const int tid = threadIdx.x;
  int n = counters[l]; if (n > CAP) n = CAP;
  float* cs = cscore + l * CAP;
  const float* cb = cbox + (size_t)l * CAP * 4;
  const int* cl = clabel + l * CAP;
  for (int j = tid; j < 1000; j += 256) {
    int o = l * 1000 + j;
    pscore[o] = -1.0f; plabel[o] = 0;
    pbox[o * 4 + 0] = 0.f; pbox[o * 4 + 1] = 0.f;
    pbox[o * 4 + 2] = 0.f; pbox[o * 4 + 3] = 0.f;
  }
  __syncthreads();
  if (n <= 1000) {
    for (int j = tid; j < n; j += 256) {
      int o = l * 1000 + j;
      pscore[o] = cs[j]; plabel[o] = cl[j];
      pbox[o * 4 + 0] = cb[j * 4 + 0]; pbox[o * 4 + 1] = cb[j * 4 + 1];
      pbox[o * 4 + 2] = cb[j * 4 + 2]; pbox[o * 4 + 3] = cb[j * 4 + 3];
    }
  } else {
    for (int it = 0; it < 1000; ++it) {
      float best = -1e30f; int bi = 0;
      for (int i = tid; i < n; i += 256) {
        float s = cs[i];
        if (s > best) { best = s; bi = i; }
      }
      sv[tid] = best; si[tid] = bi;
      __syncthreads();
      for (int off = 128; off > 0; off >>= 1) {
        if (tid < off && sv[tid + off] > sv[tid]) { sv[tid] = sv[tid + off]; si[tid] = si[tid + off]; }
        __syncthreads();
      }
      if (tid == 0) {
        int w = si[0];
        int o = l * 1000 + it;
        pscore[o] = cs[w]; plabel[o] = cl[w];
        pbox[o * 4 + 0] = cb[w * 4 + 0]; pbox[o * 4 + 1] = cb[w * 4 + 1];
        pbox[o * 4 + 2] = cb[w * 4 + 2]; pbox[o * 4 + 3] = cb[w * 4 + 3];
        cs[w] = -2.0f;
      }
      __syncthreads();
    }
  }
}

// single-block greedy class-aware NMS, 300 detections
__global__ __launch_bounds__(256)
void k_nms(const float* __restrict__ pscore, const float* __restrict__ pbox,
           const int* __restrict__ plabel, float* __restrict__ alive,
           float* __restrict__ out, int* __restrict__ out_lab) {
  __shared__ float sv[256];
  __shared__ int   si[256];
  const int tid = threadIdx.x;
  const float NEG = -1e9f;
  for (int i = tid; i < NPOOL; i += 256)
    alive[i] = (pscore[i] > 0.05f) ? pscore[i] : NEG;
  __syncthreads();
  for (int step = 0; step < 300; ++step) {
    float best = -1e30f; int bi = 0;
    for (int i = tid; i < NPOOL; i += 256) {
      float s = alive[i];
      if (s > best) { best = s; bi = i; }
    }
    sv[tid] = best; si[tid] = bi;
    __syncthreads();
    for (int off = 128; off > 0; off >>= 1) {
      if (tid < off && sv[tid + off] > sv[tid]) { sv[tid] = sv[tid + off]; si[tid] = si[tid + off]; }
      __syncthreads();
    }
    const int j = si[0];
    const bool valid = sv[0] > 0.05f;
    if (tid == 0) {
      float s = valid ? pscore[j] : 0.0f;
      out[step * 5 + 0] = s;
      out[step * 5 + 1] = valid ? pbox[j * 4 + 0] : 0.0f;
      out[step * 5 + 2] = valid ? pbox[j * 4 + 1] : 0.0f;
      out[step * 5 + 3] = valid ? pbox[j * 4 + 2] : 0.0f;
      out[step * 5 + 4] = valid ? pbox[j * 4 + 3] : 0.0f;
      out_lab[step] = valid ? plabel[j] : -1;
    }
    float oj = (float)plabel[j] * 1024.0f;
    float jx1 = pbox[j * 4 + 0] + oj, jy1 = pbox[j * 4 + 1] + oj;
    float jx2 = pbox[j * 4 + 2] + oj, jy2 = pbox[j * 4 + 3] + oj;
    float areaj = (jx2 - jx1) * (jy2 - jy1);
    __syncthreads();
    for (int i = tid; i < NPOOL; i += 256) {
      if (alive[i] > NEG + 1.0f) {
        float oi = (float)plabel[i] * 1024.0f;
        float x1 = fmaxf(jx1, pbox[i * 4 + 0] + oi);
        float y1 = fmaxf(jy1, pbox[i * 4 + 1] + oi);
        float x2 = fminf(jx2, pbox[i * 4 + 2] + oi);
        float y2 = fminf(jy2, pbox[i * 4 + 3] + oi);
        float w = fmaxf(x2 - x1, 0.0f), h = fmaxf(y2 - y1, 0.0f);
        float inter = w * h;
        float areai = (pbox[i * 4 + 2] - pbox[i * 4 + 0]) * (pbox[i * 4 + 3] - pbox[i * 4 + 1]);
        float iou = inter / (areaj + areai - inter + 1e-12f);
        if (iou > 0.5f) alive[i] = NEG;
      }
    }
    if (tid == 0) alive[j] = NEG;
    __syncthreads();
  }
}

// ---------------------------------------------------------------------------
extern "C" void kernel_launch(void* const* d_in, const int* in_sizes, int n_in,
                              void* d_out, int out_size, void* d_ws, size_t ws_size,
                              hipStream_t stream) {
  const float* feat[5] = {(const float*)d_in[0], (const float*)d_in[1], (const float*)d_in[2],
                          (const float*)d_in[3], (const float*)d_in[4]};
  const float* anchors = (const float*)d_in[5];
  const float* ctw = (const float*)d_in[6];
  const float* ctb = (const float*)d_in[7];
  const float* clw = (const float*)d_in[8];
  const float* clb = (const float*)d_in[9];
  const float* rtw = (const float*)d_in[10];
  const float* rtb = (const float*)d_in[11];
  const float* rbw = (const float*)d_in[12];
  const float* rbb = (const float*)d_in[13];

  static const int HWdim[5] = {64, 32, 16, 8, 4};
  static const int WSH[5]   = {6, 5, 4, 3, 2};
  static const int HWs[5]   = {4096, 1024, 256, 64, 16};
  static const int CUM[5]   = {0, 4096, 5120, 5376, 5440};

  char* base = (char*)d_ws;
  size_t off = 0;
  auto carve = [&](size_t bytes) { char* p = base + off; off = (off + bytes + 255) & ~(size_t)255; return p; };
  _Float16* feat_h  = (_Float16*)carve((size_t)256 * 5456 * 2);
  _Float16* t0      = (_Float16*)carve((size_t)256 * 4096 * 2);
  _Float16* t1      = (_Float16*)carve((size_t)256 * 4096 * 2);
  float*    logit_f = (float*)   carve((size_t)819 * 4096 * 4);
  float*    reg_f   = (float*)   carve((size_t)36 * 4096 * 4);
  _Float16* cls_tw  = (_Float16*)carve((size_t)1024 * 2304 * 2);
  _Float16* reg_tw  = (_Float16*)carve((size_t)1024 * 2304 * 2);
  _Float16* cls_lw  = (_Float16*)carve((size_t)819 * 2304 * 2);
  _Float16* reg_bw  = (_Float16*)carve((size_t)36 * 2304 * 2);
  int*      counters= (int*)     carve(32);
  float*    cscore  = (float*)   carve((size_t)5 * CAP * 4);
  float*    cbox    = (float*)   carve((size_t)5 * CAP * 16);
  int*      clabel  = (int*)     carve((size_t)5 * CAP * 4);
  float*    pscore  = (float*)   carve((size_t)NPOOL * 4);
  float*    pbox    = (float*)   carve((size_t)NPOOL * 16);
  int*      plabel  = (int*)     carve((size_t)NPOOL * 4);
  float*    alive   = (float*)   carve((size_t)NPOOL * 4);
  (void)ws_size; (void)n_in; (void)in_sizes; (void)out_size;

  k_pack_w<<<2048, 256, 0, stream>>>(ctw, cls_tw, 1024);
  k_pack_w<<<2048, 256, 0, stream>>>(rtw, reg_tw, 1024);
  k_pack_w<<<2048, 256, 0, stream>>>(clw, cls_lw, 819);
  k_pack_w<<<256, 256, 0, stream>>>(rbw, reg_bw, 36);
  for (int l = 0; l < 5; ++l) {
    int n = 256 * HWs[l];
    k_cvt_f16<<<(n + 255) / 256, 256, 0, stream>>>(feat[l], feat_h + (size_t)256 * CUM[l], n);
  }
  k_zero_counters<<<1, 32, 0, stream>>>(counters);

  for (int l = 0; l < 5; ++l) {
    const int H = HWdim[l], wsh = WSH[l], HW = HWs[l];
    const _Float16* fl = feat_h + (size_t)256 * CUM[l];
    dim3 g256((HW + 63) / 64, (256 + 31) / 32);
    dim3 g819((HW + 63) / 64, (819 + 31) / 32);
    dim3 g36 ((HW + 63) / 64, (36 + 31) / 32);

    const _Float16* src = fl;
    for (int layer = 0; layer < 4; ++layer) {
      _Float16* dst = (layer & 1) ? t1 : t0;
      k_conv3x3_wmma<<<g256, 256, 0, stream>>>(src, cls_tw + (size_t)layer * 256 * 2304,
                                               ctb + layer * 256, dst, nullptr, H, wsh, 256, 1);
      src = dst;
    }
    k_conv3x3_wmma<<<g819, 256, 0, stream>>>(src, cls_lw, clb, nullptr, logit_f, H, wsh, 819, 0);

    src = fl;
    for (int layer = 0; layer < 4; ++layer) {
      _Float16* dst = (layer & 1) ? t1 : t0;
      k_conv3x3_wmma<<<g256, 256, 0, stream>>>(src, reg_tw + (size_t)layer * 256 * 2304,
                                               rtb + layer * 256, dst, nullptr, H, wsh, 256, 1);
      src = dst;
    }
    k_conv3x3_wmma<<<g36, 256, 0, stream>>>(src, reg_bw, rbb, nullptr, reg_f, H, wsh, 36, 0);

    int total = HW * 819;
    int blocks = (total + 255) / 256; if (blocks > 4096) blocks = 4096;
    k_candidates<<<blocks, 256, 0, stream>>>(logit_f, reg_f, anchors, HW, 9 * CUM[l], l,
                                             counters, cscore, cbox, clabel);
  }

  k_select<<<5, 256, 0, stream>>>(counters, cscore, cbox, clabel, pscore, pbox, plabel);
  k_nms<<<1, 256, 0, stream>>>(pscore, pbox, plabel, alive,
                               (float*)d_out, (int*)d_out + 1500);
}
